// HaloAttn_40965398069447
// MI455X (gfx1250) — matrix operand order repaired and checked by reading, not compiled, not run on hardware
//
#include <hip/hip_runtime.h>
#include <hip/hip_bf16.h>

typedef __attribute__((ext_vector_type(16))) __bf16 v16bf;
typedef __attribute__((ext_vector_type(8)))  __bf16 v8bf;
typedef __attribute__((ext_vector_type(4)))  __bf16 v4bf;
typedef __attribute__((ext_vector_type(8)))  float  v8f;
typedef __attribute__((ext_vector_type(4)))  unsigned int v4u;
typedef __attribute__((ext_vector_type(8)))  int v8i;
typedef __attribute__((ext_vector_type(4)))  int v4i;

#define WMMA_BF16(A, B, C) \
  __builtin_amdgcn_wmma_f32_16x16x32_bf16(false, (A), false, (B), (short)0, (C), false, false)

// two runs of 8 bf16 at p0 and p1, each 16B-aligned (two ds_load_b128)
static __device__ inline v16bf ld_frag(const __bf16* p0, const __bf16* p1) {
  v8bf lo = *(const v8bf*)p0;
  v8bf hi = *(const v8bf*)p1;
  return __builtin_shufflevector(lo, hi, 0, 1, 2, 3, 4, 5, 6, 7,
                                         8, 9, 10, 11, 12, 13, 14, 15);
}
// 16 contiguous bf16, only 8B-aligned (four ds_load_b64) -- for TDM-packed tiles
static __device__ inline v16bf ld_frag_u8(const __bf16* p) {
  v4bf a0 = *(const v4bf*)(p);
  v4bf a1 = *(const v4bf*)(p + 4);
  v4bf a2 = *(const v4bf*)(p + 8);
  v4bf a3 = *(const v4bf*)(p + 12);
  v8bf lo = __builtin_shufflevector(a0, a1, 0, 1, 2, 3, 4, 5, 6, 7);
  v8bf hi = __builtin_shufflevector(a2, a3, 0, 1, 2, 3, 4, 5, 6, 7);
  return __builtin_shufflevector(lo, hi, 0, 1, 2, 3, 4, 5, 6, 7,
                                         8, 9, 10, 11, 12, 13, 14, 15);
}

// ---------------------------------------------------------------------------
// Kernel 1: fused Q/KV projection.  out(O=768, P=4096) = W(768,256) @ X(256,4096)
// per batch.  q -> qb bf16 (B,256,64,64); kv -> zero-padded kvp bf16 (B,512,70,70).
// ---------------------------------------------------------------------------
__global__ __launch_bounds__(256) void qkv_proj_kernel(
    const float* __restrict__ x, const float* __restrict__ wq,
    const float* __restrict__ wkv, __bf16* __restrict__ qb,
    __bf16* __restrict__ kvp) {
  constexpr int WS_STRIDE = 40, XS_STRIDE = 136;
  __shared__ __align__(32) __bf16 ws[64 * WS_STRIDE];
  __shared__ __align__(32) __bf16 xs[32 * XS_STRIDE];

  const int otile = blockIdx.x;          // 0..11 (x64 output channels)
  const int p0 = blockIdx.y * 128;       // pixel tile
  const int bz = blockIdx.z;             // batch
  const int t = threadIdx.x;
  const int lane = t & 31, wv = t >> 5;
  const int orow = wv & 3, ph = wv >> 2;
  const bool is_q = (otile < 4);         // uniform: q-tiles vs kv-tiles
  const float* wbase = is_q ? (wq + otile * 64 * 256)
                            : (wkv + (otile * 64 - 256) * 256);

  v8f acc[4] = {};
  for (int ko = 0; ko < 8; ++ko) {
    const int k0 = ko * 32;
    __syncthreads();
    {  // stage weights 64 x 32 (f32 -> bf16)
      int row = t >> 2, cb = (t & 3) * 8;
      const float* src = wbase + row * 256 + k0 + cb;
#pragma unroll
      for (int j = 0; j < 8; ++j) ws[row * WS_STRIDE + cb + j] = (__bf16)src[j];
    }
    {  // stage activations 32 x 128 (f32 -> bf16), coalesced rows
      int r = t >> 3, cb = (t & 7) * 16;
      const float* src = x + (((size_t)(bz * 256 + k0 + r)) << 12) + p0 + cb;
#pragma unroll
      for (int j = 0; j < 16; ++j) xs[r * XS_STRIDE + cb + j] = (__bf16)src[j];
    }
    __syncthreads();
    const __bf16* arow =
        ws + (orow * 16 + (lane & 15)) * WS_STRIDE + (lane >> 4) * 8;
    v16bf a = ld_frag(arow, arow + 16);
#pragma unroll
    for (int tt = 0; tt < 4; ++tt) {
      const __bf16* brow = xs + ph * 64 + tt * 16 + (lane & 15);
      v16bf bf;
#pragma unroll
      for (int s2 = 0; s2 < 16; ++s2)
        bf[s2] = brow[((s2 & 7) + (lane >> 4) * 8 + (s2 >> 3) * 16) * XS_STRIDE];
      acc[tt] = WMMA_BF16(a, bf, acc[tt]);
    }
  }
  const int Mhi = (lane >> 4) << 3;
  if (is_q) {  // uniform branch: no per-lane exec juggling
#pragma unroll
    for (int tt = 0; tt < 4; ++tt) {
#pragma unroll
      for (int r = 0; r < 8; ++r) {
        int o = otile * 64 + orow * 16 + r + Mhi;
        int p = p0 + ph * 64 + tt * 16 + (lane & 15);
        qb[(((size_t)(bz * 256 + o)) << 12) + p] = (__bf16)acc[tt][r];
      }
    }
  } else {
#pragma unroll
    for (int tt = 0; tt < 4; ++tt) {
#pragma unroll
      for (int r = 0; r < 8; ++r) {
        int och = otile * 64 - 256 + orow * 16 + r + Mhi;
        int p = p0 + ph * 64 + tt * 16 + (lane & 15);
        int hh = (p >> 6) + 3, wcol = (p & 63) + 3;
        kvp[((size_t)(bz * 512 + och) * 70 + hh) * 70 + wcol] =
            (__bf16)acc[tt][r];
      }
    }
  }
}

// ---------------------------------------------------------------------------
// Kernel 2: fused halo attention for one (batch*head, block).
// K window staged via the Tensor Data Mover (3D tile DMA); everything else in
// LDS/registers.  S = scale*Q K^T + relbias (masked), softmax, O = P V.
// ---------------------------------------------------------------------------
__global__ __launch_bounds__(256) void halo_attn_kernel(
    const __bf16* __restrict__ qb, const __bf16* __restrict__ kvp,
    const float* __restrict__ height_rel, const float* __restrict__ width_rel,
    __bf16* __restrict__ ao) {
  constexpr int QS_STRIDE = 40;    // bf16 elems
  constexpr int VS_STRIDE = 40;
  constexpr int KST_STRIDE = 196;  // TDM packs the (14,14,32) tile contiguously
  constexpr int PS_STRIDE = 232;
  constexpr int OFF_QS = 0;                 // 64*40*2   = 5120
  constexpr int OFF_VS = 5120;              // 224*40*2  = 17920
  constexpr int OFF_STATS = 23040;          // 2 * 64*2 f32 = 1024
  constexpr int OFF_A = 24064;              // overlay region, 33280 B
  constexpr int OFF_KST = OFF_A;            // 32*196*2 = 12544
  constexpr int OFF_HREL = OFF_A + 12544;   // 27*32*4 = 3456
  constexpr int OFF_WREL = OFF_HREL + 3456;
  constexpr int OFF_QH = OFF_WREL + 3456;   // 64*27*4 = 6912
  constexpr int OFF_QW = OFF_QH + 6912;
  constexpr int OFF_PS = OFF_A;             // 64*232*2 = 29696 (fits overlay)
  __shared__ __align__(32) char smem[OFF_A + 33280];  // 57344 B

  __bf16* qs = (__bf16*)(smem + OFF_QS);
  __bf16* vs = (__bf16*)(smem + OFF_VS);
  float* smax = (float*)(smem + OFF_STATS);
  float* ssum = smax + 128;
  __bf16* kst = (__bf16*)(smem + OFF_KST);
  float* hrel = (float*)(smem + OFF_HREL);
  float* wrel = (float*)(smem + OFF_WREL);
  float* qh = (float*)(smem + OFF_QH);
  float* qw = (float*)(smem + OFF_QW);
  __bf16* ps = (__bf16*)(smem + OFF_PS);

  const int t = threadIdx.x, lane = t & 31, wv = t >> 5;
  const int byi = blockIdx.x >> 3, bxi = blockIdx.x & 7;
  const int bh = blockIdx.y, b = bh >> 3, head = bh & 7;

  // ---- TDM: DMA the K window (x=14, y=14, z=32 channels) into kst ----------
  // D# per CDNA5 ISA 8.3-8.6: data_size=1 (2B), tile=(14,14,32),
  // tensor_dim0/1=(70,70), strides=(70,4900,4900), type=2.
  if (wv == 0) {
    unsigned long long gaddr = (unsigned long long)(
        kvp + ((size_t)(b * 512 + head * 64) * 4900 + (byi * 8) * 70 + bxi * 8));
    unsigned int ldsa = (unsigned int)(unsigned long long)kst;  // LDS offset
    v4u g0 = {1u, ldsa, (unsigned int)gaddr,
              (unsigned int)(gaddr >> 32) | (2u << 30)};
    v8i g1 = {(int)(1u << 16),          // data_size = 2 bytes
              (int)(70u << 16),         // tensor_dim0 = 70
              (int)(70u << 16),         // tensor_dim1 = 70
              (int)(14u << 16),         // tile_dim0 = 14
              (int)(14u | (32u << 16)), // tile_dim1 = 14, tile_dim2 = 32
              70,                       // tensor_dim0_stride = 70
              (int)(4900u << 16),       // tensor_dim1_stride = 4900 (lo16)
              0};
    v4i g2 = {32, 0, 4900, 0};          // tensor_dim2=32, dim2_stride=4900
    v4i g3 = {0, 0, 0, 0};
    v8i g4 = {0, 0, 0, 0, 0, 0, 0, 0};  // extra descriptor words (unused)
    __builtin_amdgcn_tensor_load_to_lds(g0, g1, g2, g3, g4, 0);
  }

  // ---- stage Q, V, rel tables (f32/bf16 VALU path) ----
  {  // qs[q][d], q = qy*8+qx
    int d = t >> 3, qy = t & 7;
    const __bf16* src = qb + (((size_t)(b * 256 + head * 32 + d)) << 12) +
                        (byi * 8 + qy) * 64 + bxi * 8;
#pragma unroll
    for (int qx = 0; qx < 8; ++qx) qs[(qy * 8 + qx) * QS_STRIDE + d] = src[qx];
  }
  for (int i = t; i < 224 * 32; i += 256) {  // vs[k][dv], keys padded to 224
    int k = i >> 5, dv = i & 31;
    __bf16 val = (__bf16)0.f;
    if (k < 196) {
      int wy = k / 14, wx = k % 14;
      val = kvp[((size_t)(b * 512 + head * 64 + 32 + dv) * 70 + byi * 8 + wy) *
                    70 + bxi * 8 + wx];
    }
    vs[k * VS_STRIDE + dv] = val;
  }
  for (int i = t; i < 27 * 32; i += 256) {
    hrel[i] = height_rel[i];
    wrel[i] = width_rel[i];
  }
  if (wv == 0) __builtin_amdgcn_s_wait_tensorcnt(0);  // K tile landed in LDS
  __syncthreads();

  // ---- per-query rel-bias tables: qh[q][r] = q . hrel[r], likewise qw ----
  for (int i = t; i < 64 * 27; i += 256) {
    int q = i / 27, r = i % 27;
    float sh = 0.f, sw = 0.f;
#pragma unroll
    for (int d = 0; d < 32; ++d) {
      float qv = (float)qs[q * QS_STRIDE + d];
      sh += qv * hrel[r * 32 + d];
      sw += qv * wrel[r * 32 + d];
    }
    qh[i] = sh;
    qw[i] = sw;
  }
  __syncthreads();

  // ---- S = Q K^T, tiles: wave -> (qt = wv/2, k-half = wv&1, 7 k-tiles) ----
  const int qt = wv >> 1, kh = wv & 1;
  const int Mhi = (lane >> 4) << 3, ncol = lane & 15;
  v16bf afrag;
  {
    const __bf16* arow = qs + (qt * 16 + ncol) * QS_STRIDE + (lane >> 4) * 8;
    afrag = ld_frag(arow, arow + 16);
  }
  v8f s[7];
#pragma unroll
  for (int t7 = 0; t7 < 7; ++t7) {
    // TDM-packed kst[d][key]: per-element gather down d for column key
    const __bf16* kcp = kst + (kh * 7 + t7) * 16 + ncol;
    v16bf bf;
#pragma unroll
    for (int s2 = 0; s2 < 16; ++s2)
      bf[s2] = kcp[((s2 & 7) + (lane >> 4) * 8 + (s2 >> 3) * 16) * KST_STRIDE];
    v8f z = {};
    s[t7] = WMMA_BF16(afrag, bf, z);
  }
  // bias + mask, in registers (kcol >= 196 also kills TDM overrun garbage)
#pragma unroll
  for (int t7 = 0; t7 < 7; ++t7) {
    int kcol = (kh * 7 + t7) * 16 + ncol;
#pragma unroll
    for (int r = 0; r < 8; ++r) {
      int q = qt * 16 + r + Mhi;
      float v = -1e30f;
      if (kcol < 196) {
        int wy = kcol / 14, wx = kcol % 14;
        int ky = byi * 8 + wy - 3, kx = bxi * 8 + wx - 3;
        if (ky >= 0 && ky < 64 && kx >= 0 && kx < 64) {
          int qy = q >> 3, qx = q & 7;
          v = s[t7][r] * 0.17677669529663687f +
              qh[q * 27 + (wy - qy + 13)] + qw[q * 27 + (wx - qx + 13)];
        }
      }
      s[t7][r] = v;
    }
  }
  // ---- softmax: row max/sum via shfl within 16-lane group + LDS across halves
  float rmax[8];
#pragma unroll
  for (int r = 0; r < 8; ++r) {
    float m = s[0][r];
#pragma unroll
    for (int t7 = 1; t7 < 7; ++t7) m = fmaxf(m, s[t7][r]);
    m = fmaxf(m, __shfl_xor(m, 8, 32));
    m = fmaxf(m, __shfl_xor(m, 4, 32));
    m = fmaxf(m, __shfl_xor(m, 2, 32));
    m = fmaxf(m, __shfl_xor(m, 1, 32));
    rmax[r] = m;
  }
  if (ncol == 0) {
#pragma unroll
    for (int r = 0; r < 8; ++r) smax[(qt * 16 + r + Mhi) * 2 + kh] = rmax[r];
  }
  __syncthreads();
  float rsum[8];
#pragma unroll
  for (int r = 0; r < 8; ++r) {
    int row = qt * 16 + r + Mhi;
    float m = fmaxf(smax[row * 2], smax[row * 2 + 1]);
    float sum = 0.f;
#pragma unroll
    for (int t7 = 0; t7 < 7; ++t7) {
      float e = __expf(s[t7][r] - m);
      s[t7][r] = e;
      sum += e;
    }
    sum += __shfl_xor(sum, 8, 32);
    sum += __shfl_xor(sum, 4, 32);
    sum += __shfl_xor(sum, 2, 32);
    sum += __shfl_xor(sum, 1, 32);
    rsum[r] = sum;
  }
  if (ncol == 0) {
#pragma unroll
    for (int r = 0; r < 8; ++r) ssum[(qt * 16 + r + Mhi) * 2 + kh] = rsum[r];
  }
  __syncthreads();
  // write P (bf16) into overlay region (kst/qh/qw now dead)
#pragma unroll
  for (int r = 0; r < 8; ++r) {
    int row = qt * 16 + r + Mhi;
    float inv = 1.f / (ssum[row * 2] + ssum[row * 2 + 1]);
#pragma unroll
    for (int t7 = 0; t7 < 7; ++t7) {
      int kcol = (kh * 7 + t7) * 16 + ncol;
      ps[row * PS_STRIDE + kcol] = (__bf16)(s[t7][r] * inv);
    }
  }
  __syncthreads();

  // ---- O = P V: wave -> (qt2 = wv&3, dv-tile = wv>>2), 7 K-steps of 32 ----
  const int qt2 = wv & 3, dt = wv >> 2;
  v8f o = {};
#pragma unroll
  for (int ks = 0; ks < 7; ++ks) {
    const __bf16* arow =
        ps + (qt2 * 16 + ncol) * PS_STRIDE + ks * 32 + (lane >> 4) * 8;
    v16bf a2 = ld_frag(arow, arow + 16);
    const __bf16* brow = vs + (ks * 32) * VS_STRIDE + dt * 16 + ncol;
    v16bf b2;
#pragma unroll
    for (int s2 = 0; s2 < 16; ++s2)
      b2[s2] = brow[((s2 & 7) + (lane >> 4) * 8 + (s2 >> 3) * 16) * VS_STRIDE];
    o = WMMA_BF16(a2, b2, o);
  }
#pragma unroll
  for (int r = 0; r < 8; ++r) {
    int q = qt2 * 16 + r + Mhi;
    int ch = head * 32 + dt * 16 + ncol;
    int qy = q >> 3, qx = q & 7;
    ao[(((size_t)(b * 256 + ch)) << 12) + (byi * 8 + qy) * 64 + bxi * 8 + qx] =
        (__bf16)o[r];
  }
}

// ---------------------------------------------------------------------------
// Kernel 3: output projection out(256,4096) = Wout(256,256) @ AO + b_out.
// ---------------------------------------------------------------------------
__global__ __launch_bounds__(256) void out_proj_kernel(
    const __bf16* __restrict__ ao, const float* __restrict__ w_out,
    const float* __restrict__ b_out, float* __restrict__ out) {
  constexpr int WS_STRIDE = 40, XS_STRIDE = 136;
  __shared__ __align__(32) __bf16 ws[64 * WS_STRIDE];
  __shared__ __align__(32) __bf16 xs[32 * XS_STRIDE];

  const int otile = blockIdx.x;  // 0..3
  const int p0 = blockIdx.y * 128;
  const int bz = blockIdx.z;
  const int t = threadIdx.x;
  const int lane = t & 31, wv = t >> 5;
  const int orow = wv & 3, ph = wv >> 2;

  v8f acc[4] = {};
  for (int ko = 0; ko < 8; ++ko) {
    const int k0 = ko * 32;
    __syncthreads();
    {
      int row = t >> 2, cb = (t & 3) * 8;
      const float* src = w_out + (otile * 64 + row) * 256 + k0 + cb;
#pragma unroll
      for (int j = 0; j < 8; ++j) ws[row * WS_STRIDE + cb + j] = (__bf16)src[j];
    }
    {
      int r = t >> 3, cb = (t & 7) * 16;
      const __bf16* src = ao + (((size_t)(bz * 256 + k0 + r)) << 12) + p0 + cb;
#pragma unroll
      for (int j = 0; j < 16; ++j) xs[r * XS_STRIDE + cb + j] = src[j];
    }
    __syncthreads();
    const __bf16* arow =
        ws + (orow * 16 + (lane & 15)) * WS_STRIDE + (lane >> 4) * 8;
    v16bf a = ld_frag(arow, arow + 16);
#pragma unroll
    for (int tt = 0; tt < 4; ++tt) {
      const __bf16* brow = xs + ph * 64 + tt * 16 + (lane & 15);
      v16bf bf;
#pragma unroll
      for (int s2 = 0; s2 < 16; ++s2)
        bf[s2] = brow[((s2 & 7) + (lane >> 4) * 8 + (s2 >> 3) * 16) * XS_STRIDE];
      acc[tt] = WMMA_BF16(a, bf, acc[tt]);
    }
  }
  const int Mhi = (lane >> 4) << 3;
#pragma unroll
  for (int tt = 0; tt < 4; ++tt) {
#pragma unroll
    for (int r = 0; r < 8; ++r) {
      int o = otile * 64 + orow * 16 + r + Mhi;
      int p = p0 + ph * 64 + tt * 16 + (lane & 15);
      out[(((size_t)(bz * 256 + o)) << 12) + p] = acc[tt][r] + b_out[o];
    }
  }
}

// ---------------------------------------------------------------------------
extern "C" void kernel_launch(void* const* d_in, const int* in_sizes, int n_in,
                              void* d_out, int out_size, void* d_ws,
                              size_t ws_size, hipStream_t stream) {
  const float* x = (const float*)d_in[0];
  const float* wq = (const float*)d_in[1];
  const float* wkv = (const float*)d_in[2];
  const float* height_rel = (const float*)d_in[3];
  const float* width_rel = (const float*)d_in[4];
  const float* w_out = (const float*)d_in[5];
  const float* b_out = (const float*)d_in[6];
  float* out = (float*)d_out;

  // workspace layout (bf16 intermediates, all stay resident in the 192MB L2)
  const size_t QB_BYTES = (size_t)8 * 256 * 4096 * 2;      // 16,777,216
  const size_t KVP_BYTES = (size_t)8 * 512 * 70 * 70 * 2;  // 40,140,800
  char* ws = (char*)d_ws;
  __bf16* qb = (__bf16*)(ws);
  __bf16* kvp = (__bf16*)(ws + QB_BYTES);
  __bf16* ao = (__bf16*)(ws + QB_BYTES + KVP_BYTES);

  (void)hipMemsetAsync(kvp, 0, KVP_BYTES, stream);  // zero halo padding
  qkv_proj_kernel<<<dim3(12, 32, 8), 256, 0, stream>>>(x, wq, wkv, qb, kvp);
  halo_attn_kernel<<<dim3(64, 64), 256, 0, stream>>>(qb, kvp, height_rel,
                                                     width_rel, ao);
  out_proj_kernel<<<dim3(4, 32, 8), 256, 0, stream>>>(ao, w_out, b_out, out);
}
